// DefSampler_20169166422815
// MI455X (gfx1250) — matrix operand order repaired and checked by reading, not compiled, not run on hardware
//
#include <hip/hip_runtime.h>
#include <hip/hip_bf16.h>
#include <math.h>
#include <stdint.h>

typedef __bf16 bf16;
typedef __attribute__((ext_vector_type(16))) __bf16 v16bf;
typedef __attribute__((ext_vector_type(8)))  __bf16 v8bf;
typedef __attribute__((ext_vector_type(8)))  float  v8f;

#define B_  8
#define C_  256
#define H_  64
#define W_  64
#define HW_ (H_*W_)
#define HO_ 128
#define WO_ 128
#define G_  4
#define TM1 64      // tokens per block, kernel 1
#define TM2 256     // tokens per block, kernel 2
#define LDA 264     // bf16 LDS row stride for A tiles (528B = 132 dw; 132%64=4 -> spread banks)
#define LDT 260     // f32 LDS row stride
#define LDW 272     // bf16 LDS row stride for staged weights (544B, breaks 512B aliasing)

// float -> bf16 round-to-nearest-even
__device__ inline bf16 f2bf(float f) {
  unsigned u = __float_as_uint(f);
  u += 0x7FFFu + ((u >> 16) & 1u);
  unsigned short h = (unsigned short)(u >> 16);
  return *reinterpret_cast<bf16*>(&h);
}

// ---- CDNA5 async global->LDS copy (ASYNCcnt-tracked, 16B per lane) ----
__device__ inline void async_copy_b128(unsigned lds_byte_off, const void* gp) {
  asm volatile("global_load_async_to_lds_b128 %0, %1, off"
               :: "v"(lds_byte_off), "v"(gp) : "memory");
}
__device__ inline void wait_async0() {
  asm volatile("s_wait_asynccnt 0" ::: "memory");
}

// A fragment (16x32 bf16, M rows) from LDS row-major [row][LDA]
__device__ inline v16bf load_a_frag(const bf16* lds, int m0, int k0) {
  const int lane = threadIdx.x & 31;
  const bf16* p = lds + (m0 + (lane & 15)) * LDA + k0 + ((lane >> 4) << 4);
  v8bf lo = *(const v8bf*)p;
  v8bf hi = *(const v8bf*)(p + 8);
  v16bf a;
#pragma unroll
  for (int i = 0; i < 8; ++i) { a[i] = lo[i]; a[i + 8] = hi[i]; }
  return a;
}

// B fragment (32x16) from global bf16 weights row-major [N][K=256]
__device__ inline v16bf load_b_glb(const bf16* Wb, int n0, int k0) {
  const int lane = threadIdx.x & 31;
  return *(const v16bf*)(Wb + (size_t)(n0 + (lane & 15)) * C_ + k0 + ((lane >> 4) << 4));
}

// B fragment from LDS-staged weights, padded row stride LDW
__device__ inline v16bf load_b_lds(const bf16* Wl, int n0, int k0) {
  const int lane = threadIdx.x & 31;
  return *(const v16bf*)(Wl + (n0 + (lane & 15)) * LDW + k0 + ((lane >> 4) << 4));
}

__device__ inline v8f wmma_bf16(v16bf a, v16bf b, v8f c) {
  return __builtin_amdgcn_wmma_f32_16x16x32_bf16(false, a, false, b, (short)0, c, false, false);
}

// ---------------- weight conversion ----------------
__global__ void __launch_bounds__(256)
k_cvt(const float* __restrict__ Win, const float* __restrict__ Wout,
      const float* __restrict__ Woff, const float* __restrict__ Wmask,
      bf16* __restrict__ dIn, bf16* __restrict__ dOut,
      bf16* __restrict__ dOff, bf16* __restrict__ dMask) {
  int i = blockIdx.x * 256 + threadIdx.x;
  if (i < 65536) { dIn[i] = f2bf(Win[i]); dOut[i] = f2bf(Wout[i]); }
  if (i < 8192)  { dOff[i] = f2bf(Woff[i]); dMask[i] = f2bf(Wmask[i]); }
}

// ---------------- kernel 1: token GEMM + LN + GELU + heads + offset scatter ----------------
// smem: [0)            sW  : 256 x LDW bf16 = 139264 B   (async-staged W_in)
//       [139264)       sA  : TM1 x LDA bf16 =  33792 B   (x tile, later act tile)
//       [173056)       sT  : TM1 x LDT f32  =  66560 B
#define K1_OFF_A 139264
#define K1_OFF_T 173056
#define K1_SMEM  239616

__global__ void __launch_bounds__(128)
k_token_mlp(const float* __restrict__ x,
            const bf16* __restrict__ Wb_in,  const float* __restrict__ b_in,
            const float* __restrict__ gamma, const float* __restrict__ beta,
            const bf16* __restrict__ Wb_off, const float* __restrict__ b_off,
            const bf16* __restrict__ Wb_mask,const float* __restrict__ b_mask,
            float* __restrict__ off_ws) {
  extern __shared__ __align__(32) char smem[];
  bf16*  sW = (bf16*)smem;
  bf16*  sA = (bf16*)(smem + K1_OFF_A);
  float* sT = (float*)(smem + K1_OFF_T);
  const unsigned ldsBase = (unsigned)(uintptr_t)smem;

  const int tid = threadIdx.x;
  const int bid = blockIdx.x;
  const int b   = bid >> 6;                 // 64 blocks per batch
  const int hw0 = (bid & 63) * TM1;

  // async-stage W_in (256x256 bf16) into padded LDS rows, 16B chunks
  for (int ch = tid; ch < 8192; ch += 128) {
    int n = ch >> 5;                        // 32 chunks of 16B per 512B row
    int ir = ch & 31;
    async_copy_b128(ldsBase + (unsigned)(n * (LDW * 2) + ir * 16),
                    (const char*)Wb_in + (size_t)ch * 16);
  }

  // load x tile -> bf16 LDS (coalesced along hw within a channel), overlaps async
  const float* xb = x + (size_t)b * C_ * HW_;
  for (int i = tid; i < TM1 * C_; i += 128) {
    int tok = i & (TM1 - 1);
    int c   = i >> 6;
    sA[tok * LDA + c] = f2bf(xb[(size_t)c * HW_ + hw0 + tok]);
  }
  wait_async0();
  __syncthreads();

  const int wave = tid >> 5;
  const int lane = tid & 31;
  const int m0  = wave * 16;
  const int nco = lane & 15;
  const int mro = (lane >> 4) << 3;

  // t = x-tile @ W_in^T (B frags from LDS-staged weights)
  for (int nt = 0; nt < 16; ++nt) {
    v8f acc = {};
#pragma unroll
    for (int kt = 0; kt < 8; ++kt)
      acc = wmma_bf16(load_a_frag(sA, m0, kt * 32), load_b_lds(sW, nt * 16, kt * 32), acc);
    int n = nt * 16 + nco;
    float bias = b_in[n];
#pragma unroll
    for (int i = 0; i < 8; ++i)
      sT[(m0 + mro + i) * LDT + n] = acc[i] + bias;
  }
  __syncthreads();

  // LayerNorm over channels + exact GELU; bf16 act overwrites sA
  if (tid < TM1) {
    const float* tr = sT + tid * LDT;
    float s = 0.f, s2 = 0.f;
    for (int c = 0; c < C_; ++c) { float v = tr[c]; s += v; s2 += v * v; }
    float mu   = s * (1.0f / C_);
    float var  = s2 * (1.0f / C_) - mu * mu;
    float rstd = rsqrtf(var + 1e-6f);
    for (int c = 0; c < C_; ++c) {
      float v = (tr[c] - mu) * rstd * gamma[c] + beta[c];
      v = 0.5f * v * (1.0f + erff(v * 0.70710678118f));
      sA[tid * LDA + c] = f2bf(v);
    }
  }
  __syncthreads();

  // head GEMMs (OC=32 -> 2 N-tiles each), gated combine, pixel-shuffle scatter
  v8f accO[2], accM[2];
#pragma unroll
  for (int nt = 0; nt < 2; ++nt) {
    v8f ao = {}, am = {};
#pragma unroll
    for (int kt = 0; kt < 8; ++kt) {
      v16bf a = load_a_frag(sA, m0, kt * 32);
      ao = wmma_bf16(a, load_b_glb(Wb_off,  nt * 16, kt * 32), ao);
      am = wmma_bf16(a, load_b_glb(Wb_mask, nt * 16, kt * 32), am);
    }
    accO[nt] = ao; accM[nt] = am;
  }
#pragma unroll
  for (int nt = 0; nt < 2; ++nt) {
    int oc = nt * 16 + nco;
    float bo = b_off[oc], bm = b_mask[oc];
    int g  = oc >> 3;
    int d  = (oc >> 2) & 1;
    int sy = (oc >> 1) & 1;
    int sx = oc & 1;
#pragma unroll
    for (int i = 0; i < 8; ++i) {
      int m  = m0 + mro + i;
      int hw = hw0 + m;
      int hh = hw >> 6, ww = hw & 63;
      float z   = accM[nt][i] + bm;
      float val = (accO[nt][i] + bo) / (1.0f + expf(-z));
      int ho = sy * 64 + hh, wo = sx * 64 + ww;
      off_ws[((((size_t)b * G_ + g) * 2 + d) * HO_ + ho) * WO_ + wo] = val;
    }
  }
}

// ---------------- kernel 2: bilinear gather + output GEMM (dominant) ----------------
// 256 tokens/block, 8 waves x 32 tokens, 2Mx2N register blocking.
// smem: [0)       sW  : 256 x LDW bf16 = 139264 B  (async-staged W_out)
//       [139264)  sA  : TM2 x LDA bf16 = 135168 B  (sampled act tile)
//       [274432)  sP0 : 1024 i32   (base pixel)
//       [278528)  sDx : 1024 i32
//       [282624)  sDy : 1024 i32
//       [286720)  sW4 : 1024 x 4 f32 (corner weights)  -> total 303104 B
#define K2_OFF_A  139264
#define K2_OFF_P0 274432
#define K2_OFF_DX 278528
#define K2_OFF_DY 282624
#define K2_OFF_W4 286720
#define K2_SMEM   303104

__global__ void __launch_bounds__(256)
k_sample_proj(const float* __restrict__ x,
              const float* __restrict__ off_ws,
              const bf16* __restrict__ Wb_out, const float* __restrict__ b_out,
              float* __restrict__ out) {
  extern __shared__ __align__(32) char smem[];
  bf16*  sW  = (bf16*)smem;
  bf16*  sA  = (bf16*)(smem + K2_OFF_A);
  int*   sP0 = (int*)(smem + K2_OFF_P0);
  int*   sDx = (int*)(smem + K2_OFF_DX);
  int*   sDy = (int*)(smem + K2_OFF_DY);
  float* sW4 = (float*)(smem + K2_OFF_W4);
  const unsigned ldsBase = (unsigned)(uintptr_t)smem;

  const int tid = threadIdx.x;
  const int bid = blockIdx.x;
  const int b   = bid >> 6;            // 64 blocks per batch (16384 tokens / 256)
  const int ho0 = (bid & 63) * 2;      // block covers 2 full output rows

  // async-stage W_out into LDS (overlaps param + gather stages)
  for (int ch = tid; ch < 8192; ch += 256) {
    int n = ch >> 5;
    int ir = ch & 31;
    async_copy_b128(ldsBase + (unsigned)(n * (LDW * 2) + ir * 16),
                    (const char*)Wb_out + (size_t)ch * 16);
  }

  // stage 1: bilinear params per (token, group)
  for (int i = tid; i < TM2 * G_; i += 256) {
    int tok = i >> 2, g = i & 3;
    int ho = ho0 + (tok >> 7);
    int wo = tok & 127;
    size_t obase = ((((size_t)b * G_ + g) * 2 + 0) * HO_ + ho) * WO_ + wo;
    float ox = off_ws[obase];
    float oy = off_ws[obase + (size_t)HO_ * WO_];
    float ix = ((float)wo + 0.5f + ox) * 0.5f - 0.5f;
    float iy = ((float)ho + 0.5f + oy) * 0.5f - 0.5f;
    ix = fminf(fmaxf(ix, 0.0f), 63.0f);
    iy = fminf(fmaxf(iy, 0.0f), 63.0f);
    float x0f = floorf(ix), y0f = floorf(iy);
    float wx = ix - x0f, wy = iy - y0f;
    int x0 = min(max((int)x0f, 0), 63);
    int y0 = min(max((int)y0f, 0), 63);
    sP0[i] = y0 * 64 + x0;
    sDx[i] = (x0 < 63) ? 1  : 0;
    sDy[i] = (y0 < 63) ? 64 : 0;
    sW4[i * 4 + 0] = (1.f - wx) * (1.f - wy);
    sW4[i * 4 + 1] = wx * (1.f - wy);
    sW4[i * 4 + 2] = (1.f - wx) * wy;
    sW4[i * 4 + 3] = wx * wy;
  }
  __syncthreads();

  // stage 2: gather 4 taps per (token, channel), blend, bf16 A-tile
  for (int i = tid; i < TM2 * C_; i += 256) {
    int tok = i & (TM2 - 1), c = i >> 8;
    int g = c >> 6;
    int p = (tok << 2) + g;
    const float* base = x + ((size_t)b * C_ + c) * HW_;
    int p00 = sP0[p], dx = sDx[p], dy = sDy[p];
    float v00 = base[p00];
    float v01 = base[p00 + dx];
    float v10 = base[p00 + dy];
    float v11 = base[p00 + dy + dx];
    float sv = sW4[p * 4 + 0] * v00 + sW4[p * 4 + 1] * v01 +
               sW4[p * 4 + 2] * v10 + sW4[p * 4 + 3] * v11;
    sA[tok * LDA + c] = f2bf(sv);
  }
  wait_async0();
  __syncthreads();

  // stage 3: out = sampled @ W_out^T + b_out, 2Mx2N blocked WMMA
  const int wave = tid >> 5, lane = tid & 31;
  const int m0 = wave * 32, nco = lane & 15, mro = (lane >> 4) << 3;
  for (int nt0 = 0; nt0 < 8; ++nt0) {
    v8f a00 = {}, a01 = {}, a10 = {}, a11 = {};
#pragma unroll
    for (int kt = 0; kt < 8; ++kt) {
      v16bf fa0 = load_a_frag(sA, m0,      kt * 32);
      v16bf fa1 = load_a_frag(sA, m0 + 16, kt * 32);
      v16bf fb0 = load_b_lds(sW, nt0 * 32,      kt * 32);
      v16bf fb1 = load_b_lds(sW, nt0 * 32 + 16, kt * 32);
      a00 = wmma_bf16(fa0, fb0, a00);
      a01 = wmma_bf16(fa0, fb1, a01);
      a10 = wmma_bf16(fa1, fb0, a10);
      a11 = wmma_bf16(fa1, fb1, a11);
    }
    const v8f* accs[4] = { &a00, &a01, &a10, &a11 };
#pragma unroll
    for (int t = 0; t < 4; ++t) {
      int n = nt0 * 32 + (t & 1) * 16 + nco;
      int mt = (t >> 1) * 16;
      float bias = b_out[n];
      const v8f* av = accs[t];
#pragma unroll
      for (int i = 0; i < 8; ++i) {
        int m  = m0 + mt + mro + i;
        int ho = ho0 + (m >> 7);
        int wo = m & 127;
        out[(((size_t)b * C_ + n) * HO_ + ho) * WO_ + wo] = (*av)[i] + bias;
      }
    }
  }
}

extern "C" void kernel_launch(void* const* d_in, const int* in_sizes, int n_in,
                              void* d_out, int out_size, void* d_ws, size_t ws_size,
                              hipStream_t stream) {
  const float* x      = (const float*)d_in[0];
  const float* W_in   = (const float*)d_in[1];
  const float* b_in   = (const float*)d_in[2];
  const float* gamma  = (const float*)d_in[3];
  const float* beta   = (const float*)d_in[4];
  const float* W_off  = (const float*)d_in[5];
  const float* b_off  = (const float*)d_in[6];
  const float* W_mask = (const float*)d_in[7];
  const float* b_mask = (const float*)d_in[8];
  const float* W_out  = (const float*)d_in[9];
  const float* b_out  = (const float*)d_in[10];
  float* out = (float*)d_out;

  char* ws = (char*)d_ws;
  bf16*  Wb_in   = (bf16*)(ws);
  bf16*  Wb_out  = (bf16*)(ws + 131072);
  bf16*  Wb_off  = (bf16*)(ws + 262144);
  bf16*  Wb_mask = (bf16*)(ws + 278528);
  float* off_ws  = (float*)(ws + 294912);       // 4 MB

  k_cvt<<<256, 256, 0, stream>>>(W_in, W_out, W_off, W_mask, Wb_in, Wb_out, Wb_off, Wb_mask);

  k_token_mlp<<<512, 128, K1_SMEM, stream>>>(x, Wb_in, b_in, gamma, beta,
                                             Wb_off, b_off, Wb_mask, b_mask, off_ws);

  k_sample_proj<<<512, 256, K2_SMEM, stream>>>(x, off_ws, Wb_out, b_out, out);
}
